// MSA_72662256714410
// MI455X (gfx1250) — matrix operand order repaired and checked by reading, not compiled, not run on hardware
//
#include <hip/hip_runtime.h>
#include <hip/hip_bf16.h>
#include <math.h>

// ---------------------------------------------------------------------------
// Fused 3D shifted-cube attention for MI455X (gfx1250, wave32, WMMA).
// One workgroup (256 thr = 8 waves) per 128-token cube. All GEMMs via
// v_wmma_f32_16x16x32_f16 with f32 accumulation; all intermediates in LDS.
// ---------------------------------------------------------------------------

typedef _Float16 h8  __attribute__((ext_vector_type(8)));
typedef _Float16 h16 __attribute__((ext_vector_type(16)));
typedef float    f8  __attribute__((ext_vector_type(8)));

// LDS layout (bytes). Total 320,840 <= 327,680 (320 KB per workgroup).
#define LDS_X     0        // X cube f16 [128][192]; reused as O after QKV
#define LDS_Q     49152    // Q f16 [128][192]
#define LDS_K     98304    // K f16 [128][192]
#define LDS_VT    147456   // V^T f16 [192][128]
#define LDS_W     196608   // W^T f16 [192][192] (73,728 B); overlaid by S f32 [128][128] (65,536 B)
#define LDS_P     270336   // P f16 [128][128]
#define LDS_REL   303104   // rel_table f32 [675*6]
#define LDS_LAB   319304   // int [128] shift-mask region labels
#define LDS_RSUM  319816   // float [128] softmax row sums
#define LDS_TOK   320328   // int [128] global token offsets (elements)
#define LDS_BYTES 320840

// A-fragment (16x32 f16, row-major source, row stride ld halfs).
// Lane l: M = l&15; lanes 0-15 hold K 0..7 & 16..23, lanes 16-31 hold 8..15 & 24..31.
__device__ __forceinline__ h16 ldsA(const _Float16* base, int ld, int lane) {
  const _Float16* p = base + (lane & 15) * ld + ((lane & 16) ? 8 : 0);
  h8 lo = *(const h8*)p;
  h8 hi = *(const h8*)(p + 16);
  return __builtin_shufflevector(lo, hi, 0,1,2,3,4,5,6,7,8,9,10,11,12,13,14,15);
}

// B-fragment (32x16 f16) from an N-major buffer: row = n, contiguous K, stride ld halfs.
// Lane l: N = l&15; lanes 0-15 hold K 0..7 & 16..23, lanes 16-31 hold 8..15 & 24..31.
__device__ __forceinline__ h16 ldsB(const _Float16* base, int ld, int lane) {
  const _Float16* p = base + (lane & 15) * ld + ((lane & 16) ? 8 : 0);
  h8 lo = *(const h8*)p;
  h8 hi = *(const h8*)(p + 16);
  return __builtin_shufflevector(lo, hi, 0,1,2,3,4,5,6,7,8,9,10,11,12,13,14,15);
}

__device__ __forceinline__ f8 wmma_f16(h16 a, h16 b, f8 c) {
  return __builtin_amdgcn_wmma_f32_16x16x32_f16(false, a, false, b, (short)0, c, false, false);
}

// 128x192 = [128x192]x[192x192] GEMM. A row-major f16 (ld 192), Bt N-major f16
// (ld 192). Output f16, row-major (trans==0) or transposed [n][m] (trans==1).
__device__ __forceinline__ void gemm_xw192(const _Float16* A, const _Float16* Bt,
                                           const float* __restrict__ bias,
                                           _Float16* outh, int trans,
                                           int mtile, int lane) {
  h16 af[6];
  const _Float16* abase = A + mtile * 16 * 192;
  #pragma unroll
  for (int k = 0; k < 6; ++k) af[k] = ldsA(abase + k * 32, 192, lane);
  const int mb = mtile * 16 + ((lane & 16) ? 8 : 0);
  for (int nt = 0; nt < 12; ++nt) {
    f8 acc = {};
    const _Float16* bbase = Bt + nt * 16 * 192;
    #pragma unroll
    for (int k = 0; k < 6; ++k) acc = wmma_f16(af[k], ldsB(bbase + k * 32, 192, lane), acc);
    const int n = nt * 16 + (lane & 15);
    const float bv = bias[n];
    #pragma unroll
    for (int r = 0; r < 8; ++r) {
      float v = acc[r] + bv;
      if (trans) outh[n * 128 + (mb + r)] = (_Float16)v;
      else       outh[(mb + r) * 192 + n] = (_Float16)v;
    }
  }
}

// F.normalize along token axis: per column c, scale by 1/max(||col||, 1e-12).
__device__ __forceinline__ void colnorm192(_Float16* M, int tid) {
  for (int c = tid; c < 192; c += 256) {
    float ss = 0.f;
    for (int r = 0; r < 128; ++r) { float v = (float)M[r * 192 + c]; ss += v * v; }
    float inv = 1.f / fmaxf(sqrtf(ss), 1e-12f);
    for (int r = 0; r < 128; ++r)
      M[r * 192 + c] = (_Float16)((float)M[r * 192 + c] * inv);
  }
}

__global__ __launch_bounds__(256, 1)
void cube_attn_kernel(const float* __restrict__ x,
                      const float* __restrict__ Wq, const float* __restrict__ bq,
                      const float* __restrict__ Wk, const float* __restrict__ bk,
                      const float* __restrict__ Wv, const float* __restrict__ bv,
                      const float* __restrict__ Wp, const float* __restrict__ bp,
                      const float* __restrict__ rel, const int* __restrict__ movep,
                      float* __restrict__ out) {
  extern __shared__ char smem[];
  _Float16* Xl   = (_Float16*)(smem + LDS_X);    // X, later O
  _Float16* Qb   = (_Float16*)(smem + LDS_Q);
  _Float16* Kb   = (_Float16*)(smem + LDS_K);
  _Float16* Vt   = (_Float16*)(smem + LDS_VT);
  _Float16* Wt   = (_Float16*)(smem + LDS_W);
  float*    Sb   = (float*)   (smem + LDS_W);    // overlays weight buffer
  _Float16* Pb   = (_Float16*)(smem + LDS_P);
  float*    relb = (float*)   (smem + LDS_REL);
  int*      lab  = (int*)     (smem + LDS_LAB);
  float*    rsum = (float*)   (smem + LDS_RSUM);
  int*      tok  = (int*)     (smem + LDS_TOK);

  const int tid   = threadIdx.x;
  const int lane  = tid & 31;
  const int mtile = tid >> 5;          // wave id == M-tile (8 waves, 8 tiles)

  const int blk = blockIdx.x;          // ((b*8 + cd)*8 + ch)*8 + cw
  const int bb  = blk >> 9;
  const int cd  = (blk >> 6) & 7;
  const int chh = (blk >> 3) & 7;
  const int cw  = blk & 7;

  const int mv = movep[0] ? 1 : 0;
  const int sd = mv, sh = 4 * mv, sw = 4 * mv;

  // --- token offsets (roll folded in; store goes to the same index) + labels
  if (tid < 128) {
    int t = tid;
    int td = t >> 6, th = (t >> 3) & 7, tw = t & 7;
    int pd = cd * 2 + td, ph = chh * 8 + th, pw = cw * 8 + tw;   // rolled-grid pos
    tok[t] = (((bb * 16 + ((pd + sd) & 15)) * 64 + ((ph + sh) & 63)) * 64 +
              ((pw + sw) & 63)) * 192;
    int l = 0;
    if (mv) {
      int a = (pd < 14) ? 0 : ((pd < 15) ? 1 : 2);
      int b = (ph < 56) ? 0 : ((ph < 60) ? 1 : 2);
      int c = (pw < 56) ? 0 : ((pw < 60) ? 1 : 2);
      l = a * 9 + b * 3 + c;
    }
    lab[t] = l;
  }
  for (int i = tid; i < 675 * 6; i += 256) relb[i] = rel[i];
  __syncthreads();

  // --- stage X cube (f32 -> f16): 2 threads per token, 96 ch each
  {
    int t = tid >> 1, half = tid & 1;
    const float4* src = (const float4*)(x + tok[t] + half * 96);
    _Float16* dst = Xl + t * 192 + half * 96;
    #pragma unroll
    for (int i = 0; i < 24; ++i) {
      float4 v = src[i];
      dst[i * 4 + 0] = (_Float16)v.x; dst[i * 4 + 1] = (_Float16)v.y;
      dst[i * 4 + 2] = (_Float16)v.z; dst[i * 4 + 3] = (_Float16)v.w;
    }
  }
  // --- stage Wq transposed ([n][k], contiguous K for B-fragments)
  for (int i = tid; i < 192 * 192; i += 256) {
    int k = i / 192, n = i - k * 192;
    Wt[n * 192 + k] = (_Float16)Wq[i];
  }
  __syncthreads();

  // --- Q = X*Wq + bq ; normalize over tokens
  gemm_xw192(Xl, Wt, bq, Qb, 0, mtile, lane);
  __syncthreads();
  colnorm192(Qb, tid);
  for (int i = tid; i < 192 * 192; i += 256) {
    int k = i / 192, n = i - k * 192;
    Wt[n * 192 + k] = (_Float16)Wk[i];
  }
  __syncthreads();

  // --- K = X*Wk + bk ; normalize over tokens
  gemm_xw192(Xl, Wt, bk, Kb, 0, mtile, lane);
  __syncthreads();
  colnorm192(Kb, tid);
  for (int i = tid; i < 192 * 192; i += 256) {
    int k = i / 192, n = i - k * 192;
    Wt[n * 192 + k] = (_Float16)Wv[i];
  }
  __syncthreads();

  // --- V = X*Wv + bv, stored transposed [ch][token] for the PV B-fragments
  gemm_xw192(Xl, Wt, bv, Vt, 1, mtile, lane);
  __syncthreads();

  // --- attention, one head at a time; O overwrites Xl
  const float SCALE = 0.17677669529663687f;   // 32^-0.5
  for (int h = 0; h < 6; ++h) {
    // S = Q_h * K_h^T  (dh=32 == WMMA K, single MAC per tile)
    {
      h16 aq = ldsA(Qb + mtile * 16 * 192 + h * 32, 192, lane);
      const int mb = mtile * 16 + ((lane & 16) ? 8 : 0);
      for (int nt = 0; nt < 8; ++nt) {
        f8 acc = {};
        acc = wmma_f16(aq, ldsB(Kb + nt * 16 * 192 + h * 32, 192, lane), acc);
        const int n = nt * 16 + (lane & 15);
        #pragma unroll
        for (int r = 0; r < 8; ++r) Sb[(mb + r) * 128 + n] = acc[r];
      }
    }
    __syncthreads();
    // row softmax: scale + rel-pos bias (LDS gather) + shift mask; denom deferred
    if (tid < 128) {
      int r  = tid;
      int rd = r >> 6, rh = (r >> 3) & 7, rw = r & 7;
      int labr = lab[r];
      float* srow = Sb + r * 128;
      float mx = -1e30f;
      for (int c = 0; c < 128; ++c) {
        int cdd = c >> 6, chc = (c >> 3) & 7, cwc = c & 7;
        int idx = (rd - cdd + 1) * 225 + (rh - chc + 7) * 15 + (rw - cwc + 7);
        float v = srow[c] * SCALE + relb[idx * 6 + h];
        if (lab[c] != labr) v -= 100.f;
        srow[c] = v;
        mx = fmaxf(mx, v);
      }
      float sum = 0.f;
      _Float16* prow = Pb + r * 128;
      for (int c = 0; c < 128; ++c) {
        float e = __expf(srow[c] - mx);
        sum += e;
        prow[c] = (_Float16)e;
      }
      rsum[r] = sum;
    }
    __syncthreads();
    // O_h = P * V_h  (divide by row sum in the epilogue)
    {
      h16 ap[4];
      #pragma unroll
      for (int k = 0; k < 4; ++k) ap[k] = ldsA(Pb + mtile * 16 * 128 + k * 32, 128, lane);
      const int mb = mtile * 16 + ((lane & 16) ? 8 : 0);
      for (int nt = 0; nt < 2; ++nt) {
        f8 acc = {};
        #pragma unroll
        for (int k = 0; k < 4; ++k)
          acc = wmma_f16(ap[k], ldsB(Vt + (h * 32 + nt * 16) * 128 + k * 32, 128, lane), acc);
        const int n = h * 32 + nt * 16 + (lane & 15);
        #pragma unroll
        for (int r = 0; r < 8; ++r) {
          int m = mb + r;
          Xl[m * 192 + n] = (_Float16)(acc[r] / rsum[m]);
        }
      }
    }
    __syncthreads();
  }

  // --- final projection: out = O*Wp + bp, scattered back with inverse roll
  for (int i = tid; i < 192 * 192; i += 256) {
    int k = i / 192, n = i - k * 192;
    Wt[n * 192 + k] = (_Float16)Wp[i];
  }
  __syncthreads();
  {
    h16 af[6];
    #pragma unroll
    for (int k = 0; k < 6; ++k) af[k] = ldsA(Xl + mtile * 16 * 192 + k * 32, 192, lane);
    const int mb = mtile * 16 + ((lane & 16) ? 8 : 0);
    for (int nt = 0; nt < 12; ++nt) {
      f8 acc = {};
      #pragma unroll
      for (int k = 0; k < 6; ++k)
        acc = wmma_f16(af[k], ldsB(Wt + nt * 16 * 192 + k * 32, 192, lane), acc);
      const int n = nt * 16 + (lane & 15);
      const float bias = bp[n];
      #pragma unroll
      for (int r = 0; r < 8; ++r) {
        int m = mb + r;
        out[tok[m] + n] = acc[r] + bias;
      }
    }
  }
}

extern "C" void kernel_launch(void* const* d_in, const int* in_sizes, int n_in,
                              void* d_out, int out_size, void* d_ws, size_t ws_size,
                              hipStream_t stream) {
  const float* x   = (const float*)d_in[0];
  const float* Wq  = (const float*)d_in[1];
  const float* bq  = (const float*)d_in[2];
  const float* Wk  = (const float*)d_in[3];
  const float* bk  = (const float*)d_in[4];
  const float* Wv  = (const float*)d_in[5];
  const float* bv  = (const float*)d_in[6];
  const float* Wp  = (const float*)d_in[7];
  const float* bp  = (const float*)d_in[8];
  const float* rel = (const float*)d_in[9];
  const int*   mv  = (const int*)d_in[10];
  float* out = (float*)d_out;

  (void)in_sizes; (void)n_in; (void)out_size; (void)d_ws; (void)ws_size;

  // Allow >64KB dynamic LDS (gfx1250 WGP has 320KB). Idempotent, capture-safe.
  (void)hipFuncSetAttribute((const void*)cube_attn_kernel,
                            hipFuncAttributeMaxDynamicSharedMemorySize, LDS_BYTES);

  dim3 grid(1024), block(256);   // one workgroup per cube
  hipLaunchKernelGGL(cube_attn_kernel, grid, block, LDS_BYTES, stream,
                     x, Wq, bq, Wk, bk, Wv, bv, Wp, bp, rel, mv, out);
}
